// VQEmbedding_36618891166241
// MI455X (gfx1250) — matrix operand order repaired and checked by reading, not compiled, not run on hardware
//
#include <hip/hip_runtime.h>

typedef __attribute__((ext_vector_type(16))) _Float16 v16h;
typedef __attribute__((ext_vector_type(8)))  _Float16 v8h;
typedef __attribute__((ext_vector_type(8)))  float    v8f;
typedef __attribute__((ext_vector_type(4)))  float    v4f;

#define N_TOTAL 524288
#define K_CODES 512
#define D_DIM   64
#define EH_STRIDE 72   // padded half-stride: 36 dwords -> conflict-free ds_load_b128

// Block: 256 threads (8 waves), processes 256 rows. Each wave handles two 16-row tiles.
__global__ __launch_bounds__(256)
void vq_wmma_kernel(const float* __restrict__ z,
                    const float* __restrict__ emb,
                    float* __restrict__ zq_out,
                    int* __restrict__ idx_out) {
    __shared__ _Float16 s_eh[K_CODES * EH_STRIDE]; // f16 codebook, padded rows (72 KB)
    __shared__ float    s_en[K_CODES];             // 0.5 * ||e_k||^2
    __shared__ int      s_idx[256];

    const int t = threadIdx.x;

    // ---- Stage codebook into LDS: thread t converts rows 2t, 2t+1 (128 floats) ----
    {
        const float* src = emb + t * 128;
        float n0 = 0.f, n1 = 0.f;
        #pragma unroll
        for (int c = 0; c < 16; ++c) {
            v4f x = *(const v4f*)(src + c * 8);
            v4f y = *(const v4f*)(src + c * 8 + 4);
            v8h h = { (_Float16)x.x, (_Float16)x.y, (_Float16)x.z, (_Float16)x.w,
                      (_Float16)y.x, (_Float16)y.y, (_Float16)y.z, (_Float16)y.w };
            int row = 2 * t + (c >> 3);
            int col = (c & 7) * 8;
            *(v8h*)(&s_eh[row * EH_STRIDE + col]) = h;
            float ss = x.x*x.x + x.y*x.y + x.z*x.z + x.w*x.w
                     + y.x*y.x + y.y*y.y + y.z*y.z + y.w*y.w;
            if (c < 8) n0 += ss; else n1 += ss;
        }
        s_en[2 * t]     = 0.5f * n0;
        s_en[2 * t + 1] = 0.5f * n1;
    }
    __syncthreads();

    const int wave = t >> 5;
    const int lane = t & 31;
    const int rA   = lane & 15;   // A-matrix row held by this lane
    const int hi   = lane >> 4;   // half-wave group
    const long long blockRow = (long long)blockIdx.x * 256;

    // ---- Load A fragments for two 16-row tiles (contiguous k = 16*hi + i per 32-k step, same as B) ----
    v16h a[2][2];
    #pragma unroll
    for (int tt = 0; tt < 2; ++tt) {
        const float* zrow = z + (blockRow + wave * 32 + tt * 16 + rA) * D_DIM;
        #pragma unroll
        for (int s = 0; s < 2; ++s) {
            #pragma unroll
            for (int p = 0; p < 2; ++p) {
                int base = s * 32 + hi * 16 + p * 8;
                v4f x = *(const v4f*)(zrow + base);
                v4f y = *(const v4f*)(zrow + base + 4);
                a[tt][s][p * 8 + 0] = (_Float16)x.x;
                a[tt][s][p * 8 + 1] = (_Float16)x.y;
                a[tt][s][p * 8 + 2] = (_Float16)x.z;
                a[tt][s][p * 8 + 3] = (_Float16)x.w;
                a[tt][s][p * 8 + 4] = (_Float16)y.x;
                a[tt][s][p * 8 + 5] = (_Float16)y.y;
                a[tt][s][p * 8 + 6] = (_Float16)y.z;
                a[tt][s][p * 8 + 7] = (_Float16)y.w;
            }
        }
    }

    // ---- Main loop over 32 column tiles, two per body.
    //      best[] holds packed floats: score with low 9 mantissa bits = 511 - code,
    //      so v_max gives max-score with min-index tie-break in ONE op. ----
    float best[2][8];
    #pragma unroll
    for (int tt = 0; tt < 2; ++tt)
        #pragma unroll
        for (int j = 0; j < 8; ++j) best[tt][j] = -3.0e38f;

    const int insertBase = 511 - rA;   // 511 - code = insertBase - ct*16

    for (int ct2 = 0; ct2 < 16; ++ct2) {
        const int ct0 = ct2 * 2, ct1 = ct2 * 2 + 1;
        const int code0 = ct0 * 16 + rA;
        const int code1 = ct1 * 16 + rA;
        const int ins0  = insertBase - ct0 * 16;
        const int ins1  = insertBase - ct1 * 16;

        const _Float16* bp0 = &s_eh[code0 * EH_STRIDE + hi * 16];
        const _Float16* bp1 = &s_eh[code1 * EH_STRIDE + hi * 16];
        v8h b0lo = *(const v8h*)(bp0);
        v8h b0hi = *(const v8h*)(bp0 + 8);
        v8h b1lo = *(const v8h*)(bp0 + 32);
        v8h b1hi = *(const v8h*)(bp0 + 40);
        v8h c0lo = *(const v8h*)(bp1);
        v8h c0hi = *(const v8h*)(bp1 + 8);
        v8h c1lo = *(const v8h*)(bp1 + 32);
        v8h c1hi = *(const v8h*)(bp1 + 40);
        v16h B0 = __builtin_shufflevector(b0lo, b0hi, 0,1,2,3,4,5,6,7,8,9,10,11,12,13,14,15);
        v16h B1 = __builtin_shufflevector(b1lo, b1hi, 0,1,2,3,4,5,6,7,8,9,10,11,12,13,14,15);
        v16h C0 = __builtin_shufflevector(c0lo, c0hi, 0,1,2,3,4,5,6,7,8,9,10,11,12,13,14,15);
        v16h C1 = __builtin_shufflevector(c1lo, c1hi, 0,1,2,3,4,5,6,7,8,9,10,11,12,13,14,15);
        const float en0 = s_en[code0];
        const float en1 = s_en[code1];

        #pragma unroll
        for (int tt = 0; tt < 2; ++tt) {   // B fragments reused across both M-tiles
            v8f acc0 = {};
            acc0 = __builtin_amdgcn_wmma_f32_16x16x32_f16(false, a[tt][0], false, B0,
                                                          (short)0, acc0, false, false);
            acc0 = __builtin_amdgcn_wmma_f32_16x16x32_f16(false, a[tt][1], false, B1,
                                                          (short)0, acc0, false, false);
            v8f acc1 = {};
            acc1 = __builtin_amdgcn_wmma_f32_16x16x32_f16(false, a[tt][0], false, C0,
                                                          (short)0, acc1, false, false);
            acc1 = __builtin_amdgcn_wmma_f32_16x16x32_f16(false, a[tt][1], false, C1,
                                                          (short)0, acc1, false, false);
            #pragma unroll
            for (int j = 0; j < 8; ++j) {
                float s0 = acc0[j] - en0;
                float s1 = acc1[j] - en1;
                int p0 = (__float_as_int(s0) & 0xFFFFFE00) | ins0;  // v_bfi_b32
                int p1 = (__float_as_int(s1) & 0xFFFFFE00) | ins1;
                // distinct codes -> distinct bit patterns -> max is exactly associative
                best[tt][j] = fmaxf(fmaxf(__int_as_float(p0), __int_as_float(p1)),
                                    best[tt][j]);                   // v_max3 candidate
            }
        }
    }

    // ---- Cross-lane reduce over the 16 columns: pure shfl+max (packed total order) ----
    #pragma unroll
    for (int tt = 0; tt < 2; ++tt) {
        #pragma unroll
        for (int off = 1; off <= 8; off <<= 1) {
            #pragma unroll
            for (int j = 0; j < 8; ++j) {
                best[tt][j] = fmaxf(best[tt][j], __shfl_xor(best[tt][j], off));
            }
        }
        // Lane group 'hi' holds rows hi*8 + j; lanes rA<8 publish row hi*8 + rA
        if (rA < 8) {
            float sel = best[tt][0];
            #pragma unroll
            for (int jj = 1; jj < 8; ++jj) if (rA == jj) sel = best[tt][jj];
            int code = 511 - (__float_as_int(sel) & 511);
            int lrow = wave * 32 + tt * 16 + hi * 8 + rA;
            s_idx[lrow] = code;
            idx_out[blockRow + lrow] = code;
        }
    }
    __syncthreads();

    // ---- z_q phase: z re-read is L2-hot; fully coalesced float4 traffic ----
    const v4f* z4 = (const v4f*)(z + blockRow * D_DIM);
    const v4f* e4 = (const v4f*)emb;
    v4f*       o4 = (v4f*)(zq_out + blockRow * D_DIM);
    #pragma unroll 4
    for (int i = 0; i < 16; ++i) {
        int fi  = i * 256 + t;       // 4096 float4 per block
        int row = fi >> 4;
        int c4  = fi & 15;
        int idx = s_idx[row];
        v4f zv = z4[fi];
        v4f ev = e4[idx * 16 + c4];
        v4f r;
        r.x = ev.x + (zv.x - ev.x);
        r.y = ev.y + (zv.y - ev.y);
        r.z = ev.z + (zv.z - ev.z);
        r.w = ev.w + (zv.w - ev.w);
        o4[fi] = r;
    }
}

extern "C" void kernel_launch(void* const* d_in, const int* in_sizes, int n_in,
                              void* d_out, int out_size, void* d_ws, size_t ws_size,
                              hipStream_t stream) {
    const float* z   = (const float*)d_in[0];   // [N, 64] fp32
    const float* emb = (const float*)d_in[1];   // [512, 64] fp32
    float* zq  = (float*)d_out;                              // [N, 64] fp32
    int*   idx = (int*)(zq + (size_t)N_TOTAL * D_DIM);       // [N] int32, after z_q

    vq_wmma_kernel<<<N_TOTAL / 256, 256, 0, stream>>>(z, emb, zq, idx);
}
